// Attention_1632087572898
// MI455X (gfx1250) — matrix operand order repaired and checked by reading, not compiled, not run on hardware
//
#include <hip/hip_runtime.h>
#include <hip/hip_bf16.h>
#include <math.h>

#define DD   512
#define SS   4096
#define BB   32
#define STILE 32
#define LDSTRIDE 528   // 32 rows * 528 bf16 = 33792 B, padded to avoid bank conflicts

typedef __attribute__((ext_vector_type(16))) __bf16 v16bf;
typedef __attribute__((ext_vector_type(8)))  float  v8f;

union Frag { uint4 u[2]; v16bf v; };

// ---------------------------------------------------------------------------
// Kernel 1: convert We = W1[:, :D] to bf16, row-major [k][d]
// ---------------------------------------------------------------------------
__global__ void wcvt_kernel(const float* __restrict__ W1,
                            __hip_bfloat16* __restrict__ WeBf) {
    int i = blockIdx.x * 256 + threadIdx.x;      // 0 .. 512*512-1
    int k = i >> 9;
    int d = i & 511;
    WeBf[i] = __float2bfloat16(W1[(size_t)k * (2 * DD) + d]);
}

// ---------------------------------------------------------------------------
// Kernel 2: hb[b][k] = hidden[b,:] . Wh[k,:] + b1[k]   (Wh = W1[:, D:])
// ---------------------------------------------------------------------------
__global__ void hb_kernel(const float* __restrict__ hidden,
                          const float* __restrict__ W1,
                          const float* __restrict__ b1,
                          float* __restrict__ hb) {
    int b = blockIdx.x;
    int k = threadIdx.x;                          // blockDim.x == 512
    __shared__ float hrow[DD];
    hrow[k] = hidden[(size_t)b * DD + k];
    __syncthreads();
    const float* wrow = W1 + (size_t)k * (2 * DD) + DD;
    float acc = b1[k];
    #pragma unroll 8
    for (int d = 0; d < DD; ++d) acc = fmaf(hrow[d], wrow[d], acc);
    hb[(size_t)b * DD + k] = acc;
}

// ---------------------------------------------------------------------------
// Kernel 3 (WMMA core): for an s-tile of 32 rows compute
//   e[b,s] = sum_k tanh( enc[b,s,:].We[k,:] + hb[b,k] ) * w2[k]
// 8 waves: wave = (rowgroup rg in 0..1) x (col quarter cq in 0..3)
// each wave: 16 rows x 128 cols = 8 C-tiles, K-loop over D in steps of 32.
// ---------------------------------------------------------------------------
__global__ void __launch_bounds__(256)
attn_energy_kernel(const float* __restrict__ enc,
                   const __hip_bfloat16* __restrict__ WeBf,
                   const float* __restrict__ hb,
                   const float* __restrict__ w2,
                   float* __restrict__ e_out) {
    const int b  = blockIdx.x;
    const int s0 = blockIdx.y * STILE;
    const int tid = threadIdx.x;

    __shared__ __hip_bfloat16 atile[STILE * LDSTRIDE];
    __shared__ float e_lds[STILE];

    if (tid < STILE) e_lds[tid] = 0.0f;

    // ---- stage enc tile (32 x 512 fp32) as bf16 into LDS, float4 loads ----
    const float* ebase = enc + ((size_t)b * SS + s0) * DD;
    for (int i = tid; i < STILE * DD / 4; i += 256) {
        int row  = i >> 7;            // / (512/4)
        int col4 = i & 127;
        float4 v = ((const float4*)(ebase + (size_t)row * DD))[col4];
        int o = row * LDSTRIDE + col4 * 4;
        atile[o + 0] = __float2bfloat16(v.x);
        atile[o + 1] = __float2bfloat16(v.y);
        atile[o + 2] = __float2bfloat16(v.z);
        atile[o + 3] = __float2bfloat16(v.w);
    }
    __syncthreads();

    const int wave = tid >> 5;
    const int lane = tid & 31;
    const int rg   = wave >> 2;          // row group: 0..1  -> rows rg*16..+15
    const int cq   = wave & 3;           // col quarter: 0..3 -> k offset cq*128
    const int k0w  = cq * 128;
    const int lo   = lane & 15;
    const int hi   = lane >> 4;          // 0 or 1

    v8f acc[8];
    #pragma unroll
    for (int t = 0; t < 8; ++t) acc[t] = (v8f){0,0,0,0,0,0,0,0};

    // ---- K loop over D, 16 steps of wmma_f32_16x16x32_bf16 ----
    const int arow = rg * 16 + lo;                       // M = lane%16
    const __hip_bfloat16* abase = &atile[arow * LDSTRIDE];
    for (int d0 = 0; d0 < DD; d0 += 32) {
        // A fragment 16x32 bf16: lane holds K = hi*8 + {0..7} and +16
        Frag a;
        a.u[0] = *(const uint4*)(abase + d0 + hi * 8);
        a.u[1] = *(const uint4*)(abase + d0 + hi * 8 + 16);
        #pragma unroll
        for (int t = 0; t < 8; ++t) {
            // B fragment 32x16 bf16: lane holds col n = lane%16,
            // K = hi*8 + {0..7} and +16; We row-major [n][d] -> contiguous loads
            const int n = k0w + t * 16 + lo;
            const __hip_bfloat16* bp = WeBf + (size_t)n * DD + d0 + hi * 8;
            Frag bb;
            bb.u[0] = *(const uint4*)(bp);
            bb.u[1] = *(const uint4*)(bp + 16);
            acc[t] = __builtin_amdgcn_wmma_f32_16x16x32_bf16(
                false, a.v, false, bb.v, (short)0, acc[t], false, false);
        }
    }

    // ---- epilogue: tanh(pre + hb) * w2, reduce over k ----
    float es[8];
    #pragma unroll
    for (int i = 0; i < 8; ++i) es[i] = 0.0f;

    const float* hbb = hb + (size_t)b * DD;
    #pragma unroll
    for (int t = 0; t < 8; ++t) {
        const int col = k0w + t * 16 + lo;
        const float hbv = hbb[col];
        const float w2v = w2[col];
        #pragma unroll
        for (int i = 0; i < 8; ++i) {
            // C layout: VGPR i, lane L -> M = i + 8*(L/16), N = L%16
            es[i] += tanhf(acc[t][i] + hbv) * w2v;
        }
    }
    #pragma unroll
    for (int i = 0; i < 8; ++i) {
        float v = es[i];
        v += __shfl_xor(v, 1, 32);
        v += __shfl_xor(v, 2, 32);
        v += __shfl_xor(v, 4, 32);
        v += __shfl_xor(v, 8, 32);       // stays within each 16-lane half
        if (lo == 0)
            atomicAdd(&e_lds[rg * 16 + i + hi * 8], v);
    }
    __syncthreads();

    if (tid < STILE)
        e_out[(size_t)b * SS + s0 + tid] = e_lds[tid];
}

// ---------------------------------------------------------------------------
// Kernel 4: softmax over S per batch
// ---------------------------------------------------------------------------
__global__ void softmax_kernel(const float* __restrict__ e,
                               float* __restrict__ alpha) {
    const int b = blockIdx.x;
    const int tid = threadIdx.x;
    __shared__ float red[256];
    const float* eb = e + (size_t)b * SS;

    float m = -INFINITY;
    for (int s = tid; s < SS; s += 256) m = fmaxf(m, eb[s]);
    red[tid] = m; __syncthreads();
    for (int off = 128; off > 0; off >>= 1) {
        if (tid < off) red[tid] = fmaxf(red[tid], red[tid + off]);
        __syncthreads();
    }
    m = red[0]; __syncthreads();

    float sum = 0.0f;
    for (int s = tid; s < SS; s += 256) sum += __expf(eb[s] - m);
    red[tid] = sum; __syncthreads();
    for (int off = 128; off > 0; off >>= 1) {
        if (tid < off) red[tid] += red[tid + off];
        __syncthreads();
    }
    const float inv = 1.0f / red[0];
    for (int s = tid; s < SS; s += 256)
        alpha[(size_t)b * SS + s] = __expf(eb[s] - m) * inv;
}

// ---------------------------------------------------------------------------
// Kernel 5: context[b,d] = sum_s alpha[b,s] * enc[b,s,d]
// ---------------------------------------------------------------------------
__global__ void context_kernel(const float* __restrict__ enc,
                               const float* __restrict__ alpha,
                               float* __restrict__ out) {
    const int b = blockIdx.x;
    const int d = blockIdx.y * 256 + threadIdx.x;
    __shared__ float al[256];
    const float* eb = enc + (size_t)b * SS * DD;
    float acc = 0.0f;
    for (int sb = 0; sb < SS; sb += 256) {
        __syncthreads();
        al[threadIdx.x] = alpha[(size_t)b * SS + sb + threadIdx.x];
        __syncthreads();
        #pragma unroll 8
        for (int j = 0; j < 256; ++j)
            acc = fmaf(al[j], eb[(size_t)(sb + j) * DD + d], acc);
    }
    out[(size_t)b * DD + d] = acc;
}

// ---------------------------------------------------------------------------
extern "C" void kernel_launch(void* const* d_in, const int* in_sizes, int n_in,
                              void* d_out, int out_size, void* d_ws, size_t ws_size,
                              hipStream_t stream) {
    const float* hidden = (const float*)d_in[0];  // (32, 512)
    const float* enc    = (const float*)d_in[1];  // (32, 4096, 512)
    const float* W1     = (const float*)d_in[2];  // (512, 1024)
    const float* b1     = (const float*)d_in[3];  // (512,)
    const float* w2     = (const float*)d_in[4];  // (512,)
    float* out = (float*)d_out;                   // (32, 512)

    char* ws = (char*)d_ws;
    float*          hb    = (float*)ws;                               // 64 KB
    __hip_bfloat16* WeBf  = (__hip_bfloat16*)(ws + 65536);            // 512 KB
    float*          e     = (float*)(ws + 65536 + 524288);            // 512 KB
    float*          alpha = (float*)(ws + 65536 + 524288 + 524288);   // 512 KB

    wcvt_kernel<<<(DD * DD) / 256, 256, 0, stream>>>(W1, WeBf);
    hb_kernel<<<BB, DD, 0, stream>>>(hidden, W1, b1, hb);
    attn_energy_kernel<<<dim3(BB, SS / STILE), 256, 0, stream>>>(enc, WeBf, hb, w2, e);
    softmax_kernel<<<BB, 256, 0, stream>>>(e, alpha);
    context_kernel<<<dim3(BB, DD / 256), 256, 0, stream>>>(enc, alpha, out);
}
